// PointNet2Cls_83640193122343
// MI455X (gfx1250) — matrix operand order repaired and checked by reading, not compiled, not run on hardware
//
#include <hip/hip_runtime.h>
#include <hip/hip_bf16.h>

// ---------------------------------------------------------------------------
// PointNet++ (PointNet2Cls) forward for MI455X / gfx1250, wave32 + WMMA.
// Grouped-point MLP GEMMs use V_WMMA_F32_16X16X32_F16 with weights pre-packed
// into a per-lane B-fragment layout (contiguous k, matching A) so all fragment loads are
// contiguous b128s.
// ---------------------------------------------------------------------------

#define BN_EPS 1e-5f
#define NB 32           // batch
#define NPTS 8192       // points

typedef __attribute__((ext_vector_type(16))) _Float16 v16h;
typedef __attribute__((ext_vector_type(8)))  _Float16 v8h;
typedef __attribute__((ext_vector_type(8)))  float    v8f;

// ---------------- transpose [B,3,N] -> [B,N,3] ----------------
__global__ void k_transpose_xyz(const float* __restrict__ xyz, float* __restrict__ pts) {
    int t = blockIdx.x * blockDim.x + threadIdx.x;
    const int total = NB * NPTS * 3;
    if (t >= total) return;
    int c = t % 3;
    int n = (t / 3) % NPTS;
    int b = t / (3 * NPTS);
    pts[t] = xyz[((size_t)b * 3 + c) * NPTS + n];
}

// ---------------- farthest point sampling (one block per batch) ----------------
__global__ void k_fps(const float* __restrict__ pts, int n, int npoint, int* __restrict__ out) {
    extern __shared__ float dist[];          // [n]
    __shared__ float rmax[512];
    __shared__ int   rarg[512];
    int b = blockIdx.x;
    int tid = threadIdx.x, bs = blockDim.x;
    const float* p = pts + (size_t)b * n * 3;
    for (int i = tid; i < n; i += bs) dist[i] = 1e10f;
    __syncthreads();
    int far = 0;
    for (int it = 0; it < npoint; ++it) {
        if (tid == 0) out[b * npoint + it] = far;
        float cx = p[far * 3 + 0], cy = p[far * 3 + 1], cz = p[far * 3 + 2];
        float best = -1.0f; int bi = n;
        for (int i = tid; i < n; i += bs) {
            float dx = p[i * 3 + 0] - cx;
            float dy = p[i * 3 + 1] - cy;
            float dz = p[i * 3 + 2] - cz;
            float d = dx * dx + dy * dy + dz * dz;
            float dm = fminf(dist[i], d);
            dist[i] = dm;
            if (dm > best) { best = dm; bi = i; }   // strict > keeps first max (argmax semantics)
        }
        rmax[tid] = best; rarg[tid] = bi;
        __syncthreads();
        for (int s = bs >> 1; s > 0; s >>= 1) {
            if (tid < s) {
                if (rmax[tid + s] > rmax[tid] ||
                    (rmax[tid + s] == rmax[tid] && rarg[tid + s] < rarg[tid])) {
                    rmax[tid] = rmax[tid + s]; rarg[tid] = rarg[tid + s];
                }
            }
            __syncthreads();
        }
        far = rarg[0];
        __syncthreads();
    }
}

// ---------------- gather 3-vectors: out[b,s,:] = pts[b, idx[b,s], :] ----------------
__global__ void k_gather3(const float* __restrict__ pts, const int* __restrict__ idx,
                          float* __restrict__ out, int n, int S, int total) {
    int t = blockIdx.x * blockDim.x + threadIdx.x;
    if (t >= total) return;
    int c = t % 3;
    int q = t / 3;
    int b = q / S;
    int j = idx[q];
    out[t] = pts[((size_t)b * n + j) * 3 + c];
}

// ---------------- ball query: one wave per query (b,s) ----------------
// Keeps first `nsample` in-radius indices in ascending order; pads with first.
__global__ void k_ball_query(const float* __restrict__ pts, const float* __restrict__ nxyz,
                             int* __restrict__ idx, int n, int S, int nsample, float r2) {
    int q = blockIdx.x;             // b*S + s
    int b = q / S;
    int lane = threadIdx.x & 31;
    const float* p = pts + (size_t)b * n * 3;
    float qx = nxyz[q * 3 + 0], qy = nxyz[q * 3 + 1], qz = nxyz[q * 3 + 2];
    int count = 0, first = -1;
    for (int base = 0; base < n; base += 32) {
        int j = base + lane;
        bool inr = false;
        if (j < n) {
            float dx = p[j * 3 + 0] - qx;
            float dy = p[j * 3 + 1] - qy;
            float dz = p[j * 3 + 2] - qz;
            inr = (dx * dx + dy * dy + dz * dz) <= r2;
        }
        int pref = inr ? 1 : 0;
#pragma unroll
        for (int d = 1; d < 32; d <<= 1) {
            int up = __shfl_up(pref, d, 32);
            if (lane >= d) pref += up;
        }
        int tot = __shfl(pref, 31, 32);
        if (inr) {
            int pos = count + pref - 1;
            if (pos < nsample) idx[(size_t)q * nsample + pos] = j;
        }
        if (first < 0 && tot > 0) {
            int fj = inr ? j : 0x7fffffff;
#pragma unroll
            for (int d = 16; d > 0; d >>= 1) fj = min(fj, __shfl_xor(fj, d, 32));
            first = fj;
        }
        count += tot;
        if (count >= nsample) break;       // wave-uniform
    }
    if (first < 0) first = 0;
    int start = count < nsample ? count : nsample;
    for (int k = start + lane; k < nsample; k += 32) idx[(size_t)q * nsample + k] = first;
}

// ---------------- build f16 GEMM inputs (zero-padded K) ----------------
__global__ void k_build_a1(const float* __restrict__ pts, const float* __restrict__ nxyz,
                           const int* __restrict__ idx, _Float16* __restrict__ A) {
    int row = blockIdx.x * blockDim.x + threadIdx.x;        // b*128*32 + s*32 + k
    if (row >= NB * 128 * 32) return;
    int s = (row >> 5) & 127;
    int b = row >> 12;
    int j = idx[row];
    const float* p  = pts  + ((size_t)b * NPTS + j) * 3;
    const float* c0 = nxyz + ((size_t)b * 128  + s) * 3;
    _Float16* a = A + (size_t)row * 32;
    a[0] = (_Float16)(p[0] - c0[0]);
    a[1] = (_Float16)(p[1] - c0[1]);
    a[2] = (_Float16)(p[2] - c0[2]);
    for (int c = 3; c < 32; ++c) a[c] = (_Float16)0.0f;
}

__global__ void k_build_a2(const float* __restrict__ l1xyz, const float* __restrict__ l2xyz,
                           const float* __restrict__ l1f, const int* __restrict__ idx,
                           _Float16* __restrict__ A) {
    int row = blockIdx.x * blockDim.x + threadIdx.x;        // b*64*64 + s*64 + k
    if (row >= NB * 64 * 64) return;
    int s = (row >> 6) & 63;
    int b = row >> 12;
    int j = idx[row];
    const float* p  = l1xyz + ((size_t)b * 128 + j) * 3;
    const float* c0 = l2xyz + ((size_t)b * 64  + s) * 3;
    const float* f  = l1f   + ((size_t)b * 128 + j) * 128;
    _Float16* a = A + (size_t)row * 160;
    a[0] = (_Float16)(p[0] - c0[0]);
    a[1] = (_Float16)(p[1] - c0[1]);
    a[2] = (_Float16)(p[2] - c0[2]);
    for (int c = 0; c < 128; ++c) a[3 + c] = (_Float16)f[c];
    for (int c = 131; c < 160; ++c) a[c] = (_Float16)0.0f;
}

__global__ void k_build_a3(const float* __restrict__ l2xyz, const float* __restrict__ l2f,
                           _Float16* __restrict__ A) {
    int row = blockIdx.x * blockDim.x + threadIdx.x;        // b*64 + s
    if (row >= NB * 64) return;
    const float* p = l2xyz + (size_t)row * 3;
    const float* f = l2f   + (size_t)row * 256;
    _Float16* a = A + (size_t)row * 288;
    a[0] = (_Float16)p[0];
    a[1] = (_Float16)p[1];
    a[2] = (_Float16)p[2];
    for (int c = 0; c < 256; ++c) a[3 + c] = (_Float16)f[c];
    for (int c = 259; c < 288; ++c) a[c] = (_Float16)0.0f;
}

// ---------------- misc ----------------
__global__ void k_zero(float* __restrict__ p, int n) {
    int t = blockIdx.x * blockDim.x + threadIdx.x;
    if (t < n) p[t] = 0.0f;
}

// Pack weights f32 [K,Nn] -> f16 fragment layout [Nn/16][Kpad/32][32 lanes][16],
// zero-padding K rows up to Kpad. A lane's 16 halves are the exact contents of
// its B-fragment v16h for that (n-tile, k-block): col = nt*16 + (lane&15),
// k = kblk*32 + (lane>=16 ? 16 : 0) + j   (contiguous k order, same as the A loads).
__global__ void k_pack_w_frag(const float* __restrict__ w, _Float16* __restrict__ wp,
                              int K, int Kpad, int Nn) {
    int t = blockIdx.x * blockDim.x + threadIdx.x;
    int kblks = Kpad >> 5;
    int total = (Nn >> 4) * kblks * 32 * 16;
    if (t >= total) return;
    int j    = t & 15;
    int lane = (t >> 4) & 31;
    int rest = t >> 9;                 // nt * kblks + kblk
    int nt   = rest / kblks;
    int kblk = rest - nt * kblks;
    int col  = nt * 16 + (lane & 15);
    int kk   = (kblk << 5) + ((lane >> 4) ? 16 : 0) + j;
    wp[t] = (kk < K) ? (_Float16)w[(size_t)kk * Nn + col] : (_Float16)0.0f;
}

// ---------------- WMMA GEMM + fused bias/BN/ReLU (+ optional group max-pool) -------
// 4 waves per block; each wave computes a 16x64 strip of C (4 N-tiles), reusing
// its A fragment across 4 WMMAs per 32-wide K step. All fragment loads are
// contiguous 128-bit loads.
__global__ void __launch_bounds__(128)
k_wmma_gemm(const _Float16* __restrict__ A, const _Float16* __restrict__ Wp,
            const float* __restrict__ bias, const float* __restrict__ gamma,
            const float* __restrict__ beta, const float* __restrict__ mean,
            const float* __restrict__ var,
            int M, int Kpad, int Nn,
            _Float16* __restrict__ outH, int outStride,
            float* __restrict__ outMax, int groupK) {
    int lane = threadIdx.x & 31;
    int wv   = threadIdx.x >> 5;            // wave in block -> M tile
    int hi   = lane >> 4;
    int lr   = lane & 15;
    int mt = blockIdx.y * 4 + wv;           // 16-row tile index
    int ng = blockIdx.x;                    // group of 4 N-tiles (64 cols)
    int kblks = Kpad >> 5;
    const size_t tstride = (size_t)kblks * 512;   // halves per packed N-tile

    const _Float16* ap = A + (size_t)(mt * 16 + lr) * Kpad;
    const _Float16* bp = Wp + (size_t)(ng * 4) * tstride + (size_t)lane * 16;

    v8f acc0 = {0.f,0.f,0.f,0.f,0.f,0.f,0.f,0.f};
    v8f acc1 = acc0, acc2 = acc0, acc3 = acc0;

    for (int kb = 0; kb < kblks; ++kb) {
        // A 16x32 f16 fragment: lanes 0-15 K {0..15}; lanes 16-31 K {16..31}
        int ka = (kb << 5) + (hi ? 16 : 0);
        v16h a = *(const v16h*)(ap + ka);

        const _Float16* bk = bp + (size_t)kb * 512;
        v16h b0 = *(const v16h*)(bk);
        v16h b1 = *(const v16h*)(bk + tstride);
        v16h b2 = *(const v16h*)(bk + 2 * tstride);
        v16h b3 = *(const v16h*)(bk + 3 * tstride);

        acc0 = __builtin_amdgcn_wmma_f32_16x16x32_f16(false, a, false, b0, (short)0, acc0, false, false);
        acc1 = __builtin_amdgcn_wmma_f32_16x16x32_f16(false, a, false, b1, (short)0, acc1, false, false);
        acc2 = __builtin_amdgcn_wmma_f32_16x16x32_f16(false, a, false, b2, (short)0, acc2, false, false);
        acc3 = __builtin_amdgcn_wmma_f32_16x16x32_f16(false, a, false, b3, (short)0, acc3, false, false);
    }

    // epilogue: y = relu((acc + b - mean) * gamma*rsqrt(var+eps) + beta)
    v8f accs[4] = {acc0, acc1, acc2, acc3};
#pragma unroll
    for (int t = 0; t < 4; ++t) {
        int col = (ng * 4 + t) * 16 + lr;
        float sc = gamma[col] * rsqrtf(var[col] + BN_EPS);
        float sh = (bias[col] - mean[col]) * sc + beta[col];
#pragma unroll
        for (int r = 0; r < 8; ++r) {
            int row = mt * 16 + (hi ? 8 : 0) + r;       // C/D layout (ISA 7.12.2)
            if (row >= M) continue;
            float v = fmaxf(accs[t][r] * sc + sh, 0.0f);
            if (outH)
                outH[(size_t)row * outStride + col] = (_Float16)v;
            if (outMax)   // ReLU output >= 0, so int-compare max == float max
                atomicMax((int*)outMax + ((size_t)(row / groupK) * Nn + col), __float_as_int(v));
        }
    }
}

// ---------------- small dense heads (f32 scalar) ----------------
// mode 0: BN+ReLU, 1: ReLU, 2: none, 3: sigmoid*0.6+0.4
__global__ void k_dense(const float* __restrict__ X, const float* __restrict__ W,
                        const float* __restrict__ bias,
                        const float* __restrict__ gamma, const float* __restrict__ beta,
                        const float* __restrict__ mean, const float* __restrict__ var,
                        int M, int K, int Nn, float* __restrict__ Y, int mode) {
    int t = blockIdx.x * blockDim.x + threadIdx.x;
    if (t >= M * Nn) return;
    int m = t / Nn, c = t % Nn;
    const float* x = X + (size_t)m * K;
    float s = bias[c];
    for (int k = 0; k < K; ++k) s += x[k] * W[(size_t)k * Nn + c];
    if (mode == 0) {
        float sc = gamma[c] * rsqrtf(var[c] + BN_EPS);
        s = fmaxf((s - mean[c]) * sc + beta[c], 0.0f);
    } else if (mode == 1) {
        s = fmaxf(s, 0.0f);
    } else if (mode == 3) {
        s = 1.0f / (1.0f + expf(-s)) * 0.6f + 0.4f;
    }
    Y[t] = s;
}

// ---------------------------------------------------------------------------
extern "C" void kernel_launch(void* const* d_in, const int* in_sizes, int n_in,
                              void* d_out, int out_size, void* d_ws, size_t ws_size,
                              hipStream_t stream) {
    (void)in_sizes; (void)n_in; (void)out_size; (void)ws_size;
    const float* xyz = (const float*)d_in[0];
    // d_in[1] = knn_idx (unused by the model)
    // params flattened in pytree (sorted-key) order:
    // fc1{b,beta,gamma,mean,var,w}=2..7  fc2=8..13  fc3{b,w}=14,15
    // lp1{b,w}=16,17  lp2{b,w}=18,19
    // sa1[0]=20..25 sa1[1]=26..31 sa1[2]=32..37
    // sa2[0]=38..43 sa2[1]=44..49 sa2[2]=50..55
    // sa3[0]=56..61 sa3[1]=62..67 sa3[2]=68..73
    struct BNP { const float *b, *beta, *gamma, *mean, *var, *w; };
    auto bnp = [&](int i) {
        BNP p;
        p.b = (const float*)d_in[i];         p.beta = (const float*)d_in[i + 1];
        p.gamma = (const float*)d_in[i + 2]; p.mean = (const float*)d_in[i + 3];
        p.var = (const float*)d_in[i + 4];   p.w = (const float*)d_in[i + 5];
        return p;
    };
    BNP fc1 = bnp(2), fc2 = bnp(8);
    const float* fc3_b = (const float*)d_in[14]; const float* fc3_w = (const float*)d_in[15];
    const float* lp1_b = (const float*)d_in[16]; const float* lp1_w = (const float*)d_in[17];
    const float* lp2_b = (const float*)d_in[18]; const float* lp2_w = (const float*)d_in[19];
    BNP sa1a = bnp(20), sa1b = bnp(26), sa1c = bnp(32);
    BNP sa2a = bnp(38), sa2b = bnp(44), sa2c = bnp(50);
    BNP sa3a = bnp(56), sa3b = bnp(62), sa3c = bnp(68);

    // ---- workspace carve-out ----
    char* wsp = (char*)d_ws;
    size_t off = 0;
    auto alloc = [&](size_t bytes) -> void* {
        void* p = wsp + off;
        off = (off + bytes + 255) & ~(size_t)255;
        return p;
    };
    float* pts    = (float*)alloc((size_t)NB * NPTS * 3 * 4);
    int*   fps1   = (int*)  alloc((size_t)NB * 128 * 4);
    float* l1xyz  = (float*)alloc((size_t)NB * 128 * 3 * 4);
    int*   idx1   = (int*)  alloc((size_t)NB * 128 * 32 * 4);
    float* l1f    = (float*)alloc((size_t)NB * 128 * 128 * 4);
    int*   fps2   = (int*)  alloc((size_t)NB * 64 * 4);
    float* l2xyz  = (float*)alloc((size_t)NB * 64 * 3 * 4);
    int*   idx2   = (int*)  alloc((size_t)NB * 64 * 64 * 4);
    float* l2f    = (float*)alloc((size_t)NB * 64 * 256 * 4);
    float* feat   = (float*)alloc((size_t)NB * 1024 * 4);
    float* h256   = (float*)alloc((size_t)NB * 256 * 4);
    float* h64    = (float*)alloc((size_t)NB * 64 * 4);
    float* hl64   = (float*)alloc((size_t)NB * 64 * 4);
    _Float16* wp1a = (_Float16*)alloc((size_t)32 * 64 * 2);
    _Float16* wp1b = (_Float16*)alloc((size_t)64 * 64 * 2);
    _Float16* wp1c = (_Float16*)alloc((size_t)64 * 128 * 2);
    _Float16* wp2a = (_Float16*)alloc((size_t)160 * 128 * 2);
    _Float16* wp2b = (_Float16*)alloc((size_t)128 * 128 * 2);
    _Float16* wp2c = (_Float16*)alloc((size_t)128 * 256 * 2);
    _Float16* wp3a = (_Float16*)alloc((size_t)288 * 256 * 2);
    _Float16* wp3b = (_Float16*)alloc((size_t)256 * 512 * 2);
    _Float16* wp3c = (_Float16*)alloc((size_t)512 * 1024 * 2);
    _Float16* Abuf = (_Float16*)alloc((size_t)131072 * 160 * 2);  // max A usage
    _Float16* H1   = (_Float16*)alloc((size_t)131072 * 128 * 2);  // ping
    _Float16* H2   = (_Float16*)alloc((size_t)131072 * 128 * 2);  // pong

    const int P = 131072;   // rows in SA1 (32*128*32) and SA2 (32*64*64) grouped GEMMs

    // ---- stage 0: transpose, pack weights (fragment layout), zero pooled buffers ----
    k_transpose_xyz<<<(NB * NPTS * 3 + 255) / 256, 256, 0, stream>>>(xyz, pts);
    k_pack_w_frag<<<(32 * 64 + 255) / 256, 256, 0, stream>>>(sa1a.w, wp1a, 3, 32, 64);
    k_pack_w_frag<<<(64 * 64 + 255) / 256, 256, 0, stream>>>(sa1b.w, wp1b, 64, 64, 64);
    k_pack_w_frag<<<(64 * 128 + 255) / 256, 256, 0, stream>>>(sa1c.w, wp1c, 64, 64, 128);
    k_pack_w_frag<<<(160 * 128 + 255) / 256, 256, 0, stream>>>(sa2a.w, wp2a, 131, 160, 128);
    k_pack_w_frag<<<(128 * 128 + 255) / 256, 256, 0, stream>>>(sa2b.w, wp2b, 128, 128, 128);
    k_pack_w_frag<<<(128 * 256 + 255) / 256, 256, 0, stream>>>(sa2c.w, wp2c, 128, 128, 256);
    k_pack_w_frag<<<(288 * 256 + 255) / 256, 256, 0, stream>>>(sa3a.w, wp3a, 259, 288, 256);
    k_pack_w_frag<<<(256 * 512 + 255) / 256, 256, 0, stream>>>(sa3b.w, wp3b, 256, 256, 512);
    k_pack_w_frag<<<(512 * 1024 + 255) / 256, 256, 0, stream>>>(sa3c.w, wp3c, 512, 512, 1024);
    k_zero<<<(NB * 128 * 128 + 255) / 256, 256, 0, stream>>>(l1f, NB * 128 * 128);
    k_zero<<<(NB * 64 * 256 + 255) / 256, 256, 0, stream>>>(l2f, NB * 64 * 256);
    k_zero<<<(NB * 1024 + 255) / 256, 256, 0, stream>>>(feat, NB * 1024);

    // ---- SA1 ----
    k_fps<<<NB, 512, NPTS * 4, stream>>>(pts, NPTS, 128, fps1);
    k_gather3<<<(NB * 128 * 3 + 255) / 256, 256, 0, stream>>>(pts, fps1, l1xyz, NPTS, 128, NB * 128 * 3);
    k_ball_query<<<NB * 128, 32, 0, stream>>>(pts, l1xyz, idx1, NPTS, 128, 32, 0.2f * 0.2f);
    k_build_a1<<<(P + 255) / 256, 256, 0, stream>>>(pts, l1xyz, idx1, Abuf);
    k_wmma_gemm<<<dim3(1, P / 64), 128, 0, stream>>>(Abuf, wp1a, sa1a.b, sa1a.gamma, sa1a.beta,
        sa1a.mean, sa1a.var, P, 32, 64, H1, 64, nullptr, 1);
    k_wmma_gemm<<<dim3(1, P / 64), 128, 0, stream>>>(H1, wp1b, sa1b.b, sa1b.gamma, sa1b.beta,
        sa1b.mean, sa1b.var, P, 64, 64, H2, 64, nullptr, 1);
    k_wmma_gemm<<<dim3(2, P / 64), 128, 0, stream>>>(H2, wp1c, sa1c.b, sa1c.gamma, sa1c.beta,
        sa1c.mean, sa1c.var, P, 64, 128, nullptr, 0, l1f, 32);

    // ---- SA2 ----
    k_fps<<<NB, 128, 128 * 4, stream>>>(l1xyz, 128, 64, fps2);
    k_gather3<<<(NB * 64 * 3 + 255) / 256, 256, 0, stream>>>(l1xyz, fps2, l2xyz, 128, 64, NB * 64 * 3);
    k_ball_query<<<NB * 64, 32, 0, stream>>>(l1xyz, l2xyz, idx2, 128, 64, 64, 0.4f * 0.4f);
    k_build_a2<<<(P + 255) / 256, 256, 0, stream>>>(l1xyz, l2xyz, l1f, idx2, Abuf);
    k_wmma_gemm<<<dim3(2, P / 64), 128, 0, stream>>>(Abuf, wp2a, sa2a.b, sa2a.gamma, sa2a.beta,
        sa2a.mean, sa2a.var, P, 160, 128, H1, 128, nullptr, 1);
    k_wmma_gemm<<<dim3(2, P / 64), 128, 0, stream>>>(H1, wp2b, sa2b.b, sa2b.gamma, sa2b.beta,
        sa2b.mean, sa2b.var, P, 128, 128, H2, 128, nullptr, 1);
    k_wmma_gemm<<<dim3(4, P / 64), 128, 0, stream>>>(H2, wp2c, sa2c.b, sa2c.gamma, sa2c.beta,
        sa2c.mean, sa2c.var, P, 128, 256, nullptr, 0, l2f, 64);

    // ---- SA3 (group all) ----
    const int P3 = NB * 64;   // 2048
    k_build_a3<<<(P3 + 255) / 256, 256, 0, stream>>>(l2xyz, l2f, Abuf);
    k_wmma_gemm<<<dim3(4, P3 / 64), 128, 0, stream>>>(Abuf, wp3a, sa3a.b, sa3a.gamma, sa3a.beta,
        sa3a.mean, sa3a.var, P3, 288, 256, H1, 256, nullptr, 1);
    k_wmma_gemm<<<dim3(8, P3 / 64), 128, 0, stream>>>(H1, wp3b, sa3b.b, sa3b.gamma, sa3b.beta,
        sa3b.mean, sa3b.var, P3, 256, 512, H2, 512, nullptr, 1);
    k_wmma_gemm<<<dim3(16, P3 / 64), 128, 0, stream>>>(H2, wp3c, sa3c.b, sa3c.gamma, sa3c.beta,
        sa3c.mean, sa3c.var, P3, 512, 1024, nullptr, 0, feat, 64);

    // ---- heads ----
    float* outX = (float*)d_out;          // x: [32,1]
    float* outL = (float*)d_out + 32;     // lprot: [32,3]
    k_dense<<<32, 256, 0, stream>>>(feat, fc1.w, fc1.b, fc1.gamma, fc1.beta, fc1.mean, fc1.var,
                                    NB, 1024, 256, h256, 0);
    k_dense<<<8, 256, 0, stream>>>(h256, fc2.w, fc2.b, fc2.gamma, fc2.beta, fc2.mean, fc2.var,
                                   NB, 256, 64, h64, 0);
    k_dense<<<1, 32, 0, stream>>>(h64, fc3_w, fc3_b, nullptr, nullptr, nullptr, nullptr,
                                  NB, 64, 1, outX, 3);
    k_dense<<<8, 256, 0, stream>>>(feat, lp1_w, lp1_b, nullptr, nullptr, nullptr, nullptr,
                                   NB, 1024, 64, hl64, 1);
    k_dense<<<1, 96, 0, stream>>>(hl64, lp2_w, lp2_b, nullptr, nullptr, nullptr, nullptr,
                                  NB, 64, 3, outL, 2);
}
